// CartTensorOut_61503931678829
// MI455X (gfx1250) — matrix operand order repaired and checked by reading, not compiled, not run on hardware
//
#include <hip/hip_runtime.h>
#include <hip/hip_bf16.h>

typedef __attribute__((ext_vector_type(16))) _Float16 v16h;
typedef __attribute__((ext_vector_type(8)))  _Float16 h8;
typedef __attribute__((ext_vector_type(4)))  _Float16 h4;
typedef __attribute__((ext_vector_type(8)))  float    v8f;

#define R2 0.70710678118654752440f   /* 1/sqrt(2) */
#define S2 1.41421356237309504880f   /* sqrt(2)   */
#define R3 0.57735026918962576451f   /* 1/sqrt(3) */
#define R6 0.40824829046386301637f   /* 1/sqrt(6) */

// ---------------- workspace layout ----------------
// [0, 41984)      : 41 WMMA B-fragments, each 32 lanes x 16 halves (f16)
// [65536, 77824)  : group accumulators: g0[512] then g2[512][5] (f32, 3072 floats)
#define FRAG_WG1 0    // + nt*4 + kt   (nt 0..3, kt 0..3)
#define FRAG_WG2 16   // + g*2 + kt    (g 0..8, kt 0..1)
#define FRAG_W0  34   // + kt          (kt 0..3)
#define FRAG_W1  38   // + kt          (kt 0..1)
#define FRAG_W2  40
#define NFRAG    41
#define GACC_OFF 65536

// ---------------- per-wave LDS layout (halves) ----------------
// sP   : 16 x 128   @ 0
// vP_i : 16 x 64    @ 2048 + i*1024   (i = 0..2)
// tP_m : 16 x 32    @ 5120 + m*512    (m = 0..4)
// z1   : 16 x 64    @ 7680
#define WAVE_HALVES 8704           // 17408 B per wave, x8 waves = 139264 B/block

static __device__ __forceinline__ v8f wmma_f16(v16h a, v16h b, v8f c) {
  return __builtin_amdgcn_wmma_f32_16x16x32_f16(false, a, false, b, (short)0, c,
                                                false, false);
}

// A fragment (16x32 f16, M x K): lane&15 = row M, hi = lane>>4 selects K halves.
// halves j=0..7  -> K = k0 + hi*8 + j
// halves j=8..15 -> K = k0 + 16 + hi*8 + (j-8)
// rowbase points at contiguous-K f16 data: two aligned 16-byte LDS loads.
static __device__ __forceinline__ v16h loadA_h(const _Float16* rowbase, int k0, int hi) {
  union { v16h v; h8 h[2]; } u;
  const _Float16* p = rowbase + k0 + hi * 8;
  u.h[0] = *(const h8*)(p);
  u.h[1] = *(const h8*)(p + 16);
  return u.v;
}

// A fragment from global f32 (contiguous K), converted on the fly (x_scalar only)
static __device__ __forceinline__ v16h loadA_g32(const float* rowbase, int k0, int hi) {
  v16h a;
#pragma unroll
  for (int j = 0; j < 8; ++j) {
    a[j]     = (_Float16)rowbase[k0 + hi * 8 + j];
    a[j + 8] = (_Float16)rowbase[k0 + 16 + hi * 8 + j];
  }
  return a;
}

static __device__ __forceinline__ v16h bfrag(const _Float16* frags, int idx, int lane) {
  return ((const v16h*)frags)[idx * 32 + lane];
}

// ---------------------------------------------------------------------------
// prep: fragmentize weights into the wave32 WMMA B layout (K x 16 tile:
// lane&15 = column, halves j<8 -> K = kt*32 + (lane>>4)*8 + j, j>=8 -> K = kt*32 + 16 + (lane>>4)*8 + (j-8)), zero accumulators
// ---------------------------------------------------------------------------
__global__ void prep_kernel(const float* __restrict__ W0, const float* __restrict__ W1,
                            const float* __restrict__ W2, const float* __restrict__ Wg1,
                            const float* __restrict__ Wg2, _Float16* __restrict__ frags,
                            float* __restrict__ gacc) {
  int b = blockIdx.x;
  int lane = threadIdx.x;
  if (b == NFRAG) {  // zero the group accumulators
    for (int i = lane; i < 512 * 6; i += 32) gacc[i] = 0.0f;
    return;
  }
  const float* W;
  int ncols, nt, kt;
  if (b < 16)      { W = Wg1; ncols = 64;  nt = b >> 2;        kt = b & 3; }
  else if (b < 34) { int i = b - 16; W = Wg2; ncols = 144; nt = i >> 1; kt = i & 1; }
  else if (b < 38) { W = W0; ncols = 16; nt = 0; kt = b - 34; }
  else if (b < 40) { W = W1; ncols = 16; nt = 0; kt = b - 38; }
  else             { W = W2; ncols = 16; nt = 0; kt = 0; }
  int col = nt * 16 + (lane & 15);
  int hi  = lane >> 4;
  v16h f;
#pragma unroll
  for (int j = 0; j < 16; ++j) {
    int k = kt * 32 + (j < 8 ? hi * 8 + j : 16 + hi * 8 + (j - 8));
    f[j] = (_Float16)W[k * ncols + col];
  }
  ((v16h*)frags)[b * 32 + lane] = f;
}

// ---------------------------------------------------------------------------
// main: one 16-row slab per wave, 8 waves per block (256 thr), 45 WMMAs / wave
// ---------------------------------------------------------------------------
__global__ void __launch_bounds__(256)
main_kernel(const float* __restrict__ x_scalar, const float* __restrict__ x_sph,
            const int* __restrict__ batch_index, const float* __restrict__ bg1,
            const float* __restrict__ bg2, const float* __restrict__ wpost0,
            const float* __restrict__ wpost2, const _Float16* __restrict__ frags,
            float* __restrict__ gacc) {
  extern __shared__ _Float16 smem_h[];
  const int lane = threadIdx.x & 31;
  const int wave = threadIdx.x >> 5;
  const int row0 = blockIdx.x * 128 + wave * 16;

  _Float16* wsm = smem_h + wave * WAVE_HALVES;
  _Float16* sP  = wsm;                 // 16 x 128
  _Float16* vP0 = wsm + 2048;          // 16 x 64 per component
  _Float16* tP0 = wsm + 5120;          // 16 x 32 per m
  _Float16* z1  = wsm + 7680;          // 16 x 64

  // ---- stage x_spherical slab into LDS: convert f32->f16 once, de-interleave
  //      v (stride 3) and t (stride 5) into contiguous-K planes ----
  for (int i = lane; i < 16 * 120; i += 32) {
    int r = i / 120, c4 = i - r * 120;
    int c = c4 * 4;
    float4 f = ((const float4*)(x_sph + (size_t)(row0 + r) * 480))[c4];
    float q4[4] = {f.x, f.y, f.z, f.w};
    if (c < 128) {                       // scalar block: contiguous 4-half store
      h4 hh;
      hh[0] = (_Float16)f.x; hh[1] = (_Float16)f.y;
      hh[2] = (_Float16)f.z; hh[3] = (_Float16)f.w;
      *(h4*)(sP + r * 128 + c) = hh;
    } else if (c < 320) {                // vector block: de-interleave (u, comp)
      int j = c - 128;
#pragma unroll
      for (int q = 0; q < 4; ++q) {
        int u = (j + q) / 3, comp = (j + q) - 3 * u;
        vP0[comp * 1024 + r * 64 + u] = (_Float16)q4[q];
      }
    } else {                             // tensor block: de-interleave (u, m)
      int j = c - 320;
#pragma unroll
      for (int q = 0; q < 4; ++q) {
        int u = (j + q) / 5, m = (j + q) - 5 * u;
        tP0[m * 512 + r * 32 + u] = (_Float16)q4[q];
      }
    }
  }

  const int r  = lane & 15;   // A-matrix row / D-matrix column (= h index)
  const int hi = lane >> 4;
  const int col = r;
  const v8f vzero = {0.f, 0.f, 0.f, 0.f, 0.f, 0.f, 0.f, 0.f};

  // ---- s = s_in(16x128) @ W0 ----
  v8f s = vzero;
#pragma unroll
  for (int kt = 0; kt < 4; ++kt)
    s = wmma_f16(loadA_h(sP + r * 128, kt * 32, hi),
                 bfrag(frags, FRAG_W0 + kt, lane), s);

  // ---- v_i = v_in[:,:,i](16x64) @ W1, i = 0..2 ----
  v8f vc[3];
#pragma unroll
  for (int i = 0; i < 3; ++i) {
    v8f a = vzero;
#pragma unroll
    for (int kt = 0; kt < 2; ++kt)
      a = wmma_f16(loadA_h(vP0 + i * 1024 + r * 64, kt * 32, hi),
                   bfrag(frags, FRAG_W1 + kt, lane), a);
    vc[i] = a;
  }

  // ---- t5_m = t_in[:,:,m](16x32) @ W2, m = 0..4 ----
  v8f t[5];
#pragma unroll
  for (int m = 0; m < 5; ++m)
    t[m] = wmma_f16(loadA_h(tP0 + m * 512 + r * 32, 0, hi),
                    bfrag(frags, FRAG_W2, lane), vzero);

  // ---- gating MLP stage 1: silu(x_scalar @ Wg1 + bg1) -> z1 (LDS f16) ----
  const float* Xrow = x_scalar + (size_t)(row0 + r) * 128;
#pragma unroll
  for (int nt = 0; nt < 4; ++nt) {
    v8f acc = vzero;
#pragma unroll
    for (int kt = 0; kt < 4; ++kt)
      acc = wmma_f16(loadA_g32(Xrow, kt * 32, hi),
                     bfrag(frags, FRAG_WG1 + nt * 4 + kt, lane), acc);
    float bb = bg1[nt * 16 + col];
#pragma unroll
    for (int e = 0; e < 8; ++e) {
      float x  = acc[e] + bb;
      float sv = x / (1.0f + __expf(-x));           // silu
      z1[(e + 8 * hi) * 64 + nt * 16 + col] = (_Float16)sv;
    }
  }

  // A fragments of z1 for the second MLP GEMM (wave-private LDS, in-order DS)
  v16h aw0 = loadA_h(z1 + r * 64, 0, hi);
  v16h aw1 = loadA_h(z1 + r * 64, 32, hi);

  auto wtile = [&](int g) -> v8f {   // w[:, g, :] tile, with bg2 bias
    v8f acc = vzero;
    acc = wmma_f16(aw0, bfrag(frags, FRAG_WG2 + g * 2 + 0, lane), acc);
    acc = wmma_f16(aw1, bfrag(frags, FRAG_WG2 + g * 2 + 1, lane), acc);
    float bb = bg2[g * 16 + col];
#pragma unroll
    for (int e = 0; e < 8; ++e) acc[e] += bb;
    return acc;
  };

  const float wp00 = wpost0[0], wp01 = wpost0[1], wp02 = wpost0[2];
  const float wp20 = wpost2[0], wp21 = wpost2[1], wp22 = wpost2[2], wp23 = wpost2[3];

  v8f o0p = vzero;
  v8f o2p[5] = {vzero, vzero, vzero, vzero, vzero};

  {
    v8f w0t = wtile(0);
#pragma unroll
    for (int e = 0; e < 8; ++e) o0p[e] += wp00 * w0t[e] * s[e] * s[e];
  }
  {
    v8f w1t = wtile(1);
#pragma unroll
    for (int e = 0; e < 8; ++e) {
      float c = wp20 * w1t[e] * s[e];
#pragma unroll
      for (int m = 0; m < 5; ++m) o2p[m][e] += c * t[m][e];
    }
  }
  {
    v8f w2t = wtile(2);
#pragma unroll
    for (int e = 0; e < 8; ++e) {
      float nv = vc[0][e] * vc[0][e] + vc[1][e] * vc[1][e] + vc[2][e] * vc[2][e];
      o0p[e] += wp01 * w2t[e] * nv;
    }
  }
  {
    v8f w4t = wtile(4);
#pragma unroll
    for (int e = 0; e < 8; ++e) {
      float v0 = vc[0][e], v1 = vc[1][e], v2 = vc[2][e];
      float c = wp21 * w4t[e];
      o2p[0][e] += c * (S2 * v0 * v2);
      o2p[1][e] += c * (S2 * v0 * v1);
      o2p[2][e] += c * (R6 * (2.f * v1 * v1 - v0 * v0 - v2 * v2));
      o2p[3][e] += c * (S2 * v1 * v2);
      o2p[4][e] += c * (R2 * (v2 * v2 - v0 * v0));
    }
  }
  {
    v8f w5t = wtile(5);
#pragma unroll
    for (int e = 0; e < 8; ++e) {
      float c = wp22 * w5t[e] * s[e];
#pragma unroll
      for (int m = 0; m < 5; ++m) o2p[m][e] += c * t[m][e];
    }
  }
  {
    v8f w6t = wtile(6);
#pragma unroll
    for (int e = 0; e < 8; ++e) {
      float nt2 = t[0][e] * t[0][e] + t[1][e] * t[1][e] + t[2][e] * t[2][e] +
                  t[3][e] * t[3][e] + t[4][e] * t[4][e];
      o0p[e] += wp02 * w6t[e] * nt2;
    }
  }
  {
    v8f w8t = wtile(8);
#pragma unroll
    for (int e = 0; e < 8; ++e) {
      float t0 = t[0][e], t1 = t[1][e], t2 = t[2][e], t3 = t[3][e], t4 = t[4][e];
      float T00 = -R6 * t2 - R2 * t4;
      float T11 = 2.f * R6 * t2;
      float T22 = -R6 * t2 + R2 * t4;
      float T01 = R2 * t1, T02 = R2 * t0, T12 = R2 * t3;
      float P00 = T00 * T00 + T01 * T01 + T02 * T02;
      float P11 = T01 * T01 + T11 * T11 + T12 * T12;
      float P22 = T02 * T02 + T12 * T12 + T22 * T22;
      float P01 = T01 * (T00 + T11) + T02 * T12;
      float P02 = T02 * (T00 + T22) + T01 * T12;
      float P12 = T12 * (T11 + T22) + T01 * T02;
      float c = wp23 * w8t[e];
      o2p[0][e] += c * (S2 * P02);
      o2p[1][e] += c * (S2 * P01);
      o2p[2][e] += c * (R6 * (2.f * P11 - P00 - P22));
      o2p[3][e] += c * (S2 * P12);
      o2p[4][e] += c * (R2 * (P22 - P00));
    }
  }

  // ---- reduce over the 16 h-columns (lanes 0..15 / 16..31 independently) ----
#pragma unroll
  for (int off = 1; off < 16; off <<= 1) {
#pragma unroll
    for (int e = 0; e < 8; ++e) {
      o0p[e] += __shfl_xor(o0p[e], off, 32);
#pragma unroll
      for (int m = 0; m < 5; ++m) o2p[m][e] += __shfl_xor(o2p[m][e], off, 32);
    }
  }

  // lane 0 holds rows 0..7 (elements), lane 16 holds rows 8..15
  if ((lane & 15) == 0) {
#pragma unroll
    for (int e = 0; e < 8; ++e) {
      int row = row0 + e + 8 * hi;
      int b = batch_index[row];
      atomicAdd(&gacc[b], o0p[e]);
#pragma unroll
      for (int m = 0; m < 5; ++m) atomicAdd(&gacc[512 + b * 5 + m], o2p[m][e]);
    }
  }
}

// ---------------------------------------------------------------------------
// finalize: res_sph -> Q_COB -> CART_PERM (antisymmetric coeffs are zero)
// ---------------------------------------------------------------------------
__global__ void final_kernel(const float* __restrict__ gacc, float* __restrict__ out) {
  int g = blockIdx.x * blockDim.x + threadIdx.x;
  if (g >= 512) return;
  float g0 = gacc[g];
  float m0 = gacc[512 + g * 5 + 0];
  float m1 = gacc[512 + g * 5 + 1];
  float m2 = gacc[512 + g * 5 + 2];
  float m3 = gacc[512 + g * 5 + 3];
  float m4 = gacc[512 + g * 5 + 4];
  float M[3][3];
  M[0][0] = g0 * R3 - R6 * m2 - R2 * m4;
  M[1][1] = g0 * R3 + 2.f * R6 * m2;
  M[2][2] = g0 * R3 - R6 * m2 + R2 * m4;
  M[0][1] = M[1][0] = R2 * m1;
  M[0][2] = M[2][0] = R2 * m0;
  M[1][2] = M[2][1] = R2 * m3;
  const int perm[3] = {2, 0, 1};
#pragma unroll
  for (int i = 0; i < 3; ++i)
#pragma unroll
    for (int j = 0; j < 3; ++j)
      out[g * 9 + i * 3 + j] = M[perm[i]][perm[j]];
}

extern "C" void kernel_launch(void* const* d_in, const int* in_sizes, int n_in,
                              void* d_out, int out_size, void* d_ws, size_t ws_size,
                              hipStream_t stream) {
  (void)in_sizes; (void)n_in; (void)out_size; (void)ws_size;
  const float* x_scalar    = (const float*)d_in[0];
  const float* x_spherical = (const float*)d_in[1];
  /* d_in[2] = coord : unused by the reference */
  const int*   batch_index = (const int*)d_in[3];
  const float* W0  = (const float*)d_in[4];
  const float* W1  = (const float*)d_in[5];
  const float* W2  = (const float*)d_in[6];
  const float* Wg1 = (const float*)d_in[7];
  const float* bg1 = (const float*)d_in[8];
  const float* Wg2 = (const float*)d_in[9];
  const float* bg2 = (const float*)d_in[10];
  const float* wpost0 = (const float*)d_in[11];
  const float* wpost2 = (const float*)d_in[12];

  _Float16* frags = (_Float16*)d_ws;
  float*    gacc  = (float*)((char*)d_ws + GACC_OFF);
  float*    out   = (float*)d_out;

  prep_kernel<<<NFRAG + 1, 32, 0, stream>>>(W0, W1, W2, Wg1, Wg2, frags, gacc);

  const int N = 131072;
  const size_t lds_bytes = 8 * WAVE_HALVES * sizeof(_Float16);  // 139264 B
  main_kernel<<<N / 128, 256, lds_bytes, stream>>>(x_scalar, x_spherical, batch_index,
                                                   bg1, bg2, wpost0, wpost2, frags, gacc);

  final_kernel<<<2, 256, 0, stream>>>(gacc, out);
}
